// UTDEModule_59708635349352
// MI455X (gfx1250) — matrix operand order repaired and checked by reading, not compiled, not run on hardware
//
#include <hip/hip_runtime.h>
#include <hip/hip_bf16.h>
#include <math.h>

// Problem dims (fixed by reference)
#define DM 64
#define LOBS 256
#define NALPHA 256
#define DH 256
#define DV 64
#define NH 8

typedef __attribute__((ext_vector_type(16))) _Float16 v16h;
typedef __attribute__((ext_vector_type(8)))  float    v8f;

// ---- WMMA operand gathers, contiguous k order (k = 16*(lane/16) + i) used for both A and B ----
// 16-bit A 16x32 (MxK): lane m = lane%16; elem i -> k = i+16g, g=lane/16
static __device__ __forceinline__ int a_klocal(int i, int g) {
  return i + 16 * g;
}

// A operand with theta computed on the fly: theta[k] = (k==0) ? w*tau+phi : sin(w*tau+phi)
// __sinf -> v_sin_f32 (hardware range reduction covers |arg| here, ~240 revolutions max)
static __device__ __forceinline__ v16h theta_A(const float* __restrict__ w,
                                               const float* __restrict__ phi,
                                               float tau, int k0, int lane) {
  int g = (lane >> 4) & 1;
  v16h a;
#pragma unroll
  for (int i = 0; i < 16; ++i) {
    int k = k0 + a_klocal(i, g);
    float v = w[k] * tau + phi[k];
    a[i] = (_Float16)((k == 0) ? v : __sinf(v));
  }
  return a;
}

// A operand gathered from f16 [M][K] row-major array
static __device__ __forceinline__ v16h load_A(const _Float16* __restrict__ base,
                                              int lda, int m0, int k0, int lane) {
  int g = (lane >> 4) & 1;
  int m = m0 + (lane & 15);
  v16h a;
#pragma unroll
  for (int i = 0; i < 16; ++i) a[i] = base[m * lda + k0 + a_klocal(i, g)];
  return a;
}

// B operand (K x N): lanes 0-15 hold K=k0..k0+15, lanes 16-31 hold K=k0+16..k0+31,
// column n = lane%16. Source stored [N][K] row-major.
static __device__ __forceinline__ v16h load_B_f16(const _Float16* __restrict__ baseNK,
                                                  int ldk, int n0, int k0, int lane) {
  int g = (lane >> 4) & 1;
  int n = n0 + (lane & 15);
  v16h b;
#pragma unroll
  for (int i = 0; i < 16; ++i) b[i] = baseNK[n * ldk + k0 + 16 * g + i];
  return b;
}

static __device__ __forceinline__ v16h load_B_f32(const float* __restrict__ baseNK,
                                                  int ldk, int n0, int k0, int lane) {
  int g = (lane >> 4) & 1;
  int n = n0 + (lane & 15);
  v16h b;
#pragma unroll
  for (int i = 0; i < 16; ++i) b[i] = (_Float16)baseNK[n * ldk + k0 + 16 * g + i];
  return b;
}

#define WMMA_F16(A, B, C) \
  __builtin_amdgcn_wmma_f32_16x16x32_f16(false, (A), false, (B), (short)0, (C), false, false)

// ---------------------------------------------------------------------------
// Kernel 1: imputation (scatter last-write-wins into ALPHA grid + forward fill)
// ---------------------------------------------------------------------------
__global__ void utde_impute(const float* __restrict__ x_ts,
                            const float* __restrict__ t_ts,
                            const float* __restrict__ gmeans,
                            float* __restrict__ regular) {
  int m = threadIdx.x;
  if (m >= DM) return;
  float* row = regular + m * NALPHA;
  const float kNaN = __int_as_float(0x7fc00000);
  for (int a = 0; a < NALPHA; ++a) row[a] = kNaN;
  for (int l = 0; l < LOBS; ++l) {
    float xv = x_ts[m * LOBS + l];
    float tv = t_ts[m * LOBS + l];
    if ((xv == xv) && (tv >= 0.0f)) {          // valid observation
      int idx = (int)tv;                        // truncation, matches reference
      if (idx < NALPHA) row[idx] = xv;          // last write wins (serial loop)
    }
  }
  float cur = gmeans[m];
  for (int a = 0; a < NALPHA; ++a) {
    float v = row[a];
    if (v == v) cur = v;                        // forward fill
    row[a] = cur;
  }
}

// ---------------------------------------------------------------------------
// Kernel 2: q = (theta_a @ wq^T + wq_b) / sqrt(DV), stored f16 [h][a][e]
// ---------------------------------------------------------------------------
__global__ void utde_qprep(const float* __restrict__ t2v_w,
                           const float* __restrict__ t2v_phi,
                           const float* __restrict__ wq_w,
                           const float* __restrict__ wq_b,
                           _Float16* __restrict__ qout) {
  __shared__ float wsh[DV], psh[DV];
  int h = blockIdx.x;
  int tid = threadIdx.x;
  if (tid < DV) { wsh[tid] = t2v_w[h * DV + tid]; psh[tid] = t2v_phi[h * DV + tid]; }
  __syncthreads();
  int wave = tid >> 5, lane = tid & 31;
  int g = (lane >> 4) & 1, n = lane & 15;
  const float* wq = wq_w + h * DV * DV;   // [e][d] == B stored [N][K]
#pragma unroll 1
  for (int mt2 = 0; mt2 < 2; ++mt2) {
    int mt = wave * 2 + mt2;              // a-tile (16 alpha rows)
    float tau = (float)(mt * 16 + n);     // alpha grid point for this lane's row
    v16h a0 = theta_A(wsh, psh, tau, 0, lane);
    v16h a1 = theta_A(wsh, psh, tau, 32, lane);
#pragma unroll 1
    for (int nt = 0; nt < 4; ++nt) {
      v16h b0 = load_B_f32(wq, DV, nt * 16, 0, lane);
      v16h b1 = load_B_f32(wq, DV, nt * 16, 32, lane);
      v8f acc = {};
      acc = WMMA_F16(a0, b0, acc);
      acc = WMMA_F16(a1, b1, acc);
      int e = nt * 16 + n;
      float bias = wq_b[h * DV + e];
#pragma unroll
      for (int v = 0; v < 8; ++v) {
        int ar = mt * 16 + v + 8 * g;     // C/D row = v + 8*(lane/16)
        qout[(h * NALPHA + ar) * DV + e] = (_Float16)((acc[v] + bias) * 0.125f);
      }
    }
  }
}

// ---------------------------------------------------------------------------
// Kernel 3: attention per (head h, feature j): k-proj (WMMA) -> scores (WMMA)
//           -> ONLINE register softmax -> interp[h][j][a]
// ---------------------------------------------------------------------------
__global__ void utde_attn(const float* __restrict__ x_ts,
                          const float* __restrict__ t_ts,
                          const float* __restrict__ t2v_w,
                          const float* __restrict__ t2v_phi,
                          const float* __restrict__ wk_w,
                          const float* __restrict__ wk_b,
                          const _Float16* __restrict__ qf16,
                          float* __restrict__ interp) {
  __shared__ float wsh[DV], psh[DV];
  __shared__ float trow[LOBS], xrow[LOBS], vmask[LOBS];
  __shared__ _Float16 kmat[LOBS * DV];    // k [l][e], 32 KB
  int h = blockIdx.x, j = blockIdx.y;
  int tid = threadIdx.x;
  if (tid < DV) { wsh[tid] = t2v_w[h * DV + tid]; psh[tid] = t2v_phi[h * DV + tid]; }
  {
    float xv = x_ts[j * LOBS + tid];
    float tv = t_ts[j * LOBS + tid];
    bool valid = (xv == xv) && (tv >= 0.0f);
    trow[tid]  = valid ? tv : 0.0f;
    xrow[tid]  = valid ? xv : 0.0f;
    vmask[tid] = valid ? 1.0f : 0.0f;
  }
  __syncthreads();
  int wave = tid >> 5, lane = tid & 31;
  int g = (lane >> 4) & 1, n = lane & 15;
  const float* wk = wk_w + h * DV * DV;   // [e][d] == B stored [N][K]

  // ---- k = theta_o @ wk^T + wk_b  ->  kmat (f16) ----
#pragma unroll 1
  for (int lt2 = 0; lt2 < 2; ++lt2) {
    int lt = wave * 2 + lt2;
    float tau = trow[lt * 16 + n];
    v16h a0 = theta_A(wsh, psh, tau, 0, lane);
    v16h a1 = theta_A(wsh, psh, tau, 32, lane);
#pragma unroll 1
    for (int nt = 0; nt < 4; ++nt) {
      v16h b0 = load_B_f32(wk, DV, nt * 16, 0, lane);
      v16h b1 = load_B_f32(wk, DV, nt * 16, 32, lane);
      v8f acc = {};
      acc = WMMA_F16(a0, b0, acc);
      acc = WMMA_F16(a1, b1, acc);
      int e = nt * 16 + n;
      float bias = wk_b[h * DV + e];
#pragma unroll
      for (int v = 0; v < 8; ++v)
        kmat[(lt * 16 + v + 8 * g) * DV + e] = (_Float16)(acc[v] + bias);
    }
  }
  __syncthreads();

  // ---- scores (q already carries 1/sqrt(DV)) + online softmax + interp ----
  const _Float16* qbase = qf16 + h * NALPHA * DV;
#pragma unroll 1
  for (int at2 = 0; at2 < 2; ++at2) {
    int at = wave * 2 + at2;              // a-tile of 16 alpha rows
    v16h qa0 = load_A(qbase, DV, at * 16, 0, lane);
    v16h qa1 = load_A(qbase, DV, at * 16, 32, lane);
    // per-lane running softmax state for 8 rows (C/D rows v + 8g)
    float mx[8], se[8], sx[8];
#pragma unroll
    for (int v = 0; v < 8; ++v) { mx[v] = -3.0e38f; se[v] = 0.0f; sx[v] = 0.0f; }
#pragma unroll 1
    for (int lt = 0; lt < 16; ++lt) {
      v16h b0 = load_B_f16(kmat, DV, lt * 16, 0, lane);
      v16h b1 = load_B_f16(kmat, DV, lt * 16, 32, lane);
      v8f acc = {};
      acc = WMMA_F16(qa0, b0, acc);
      acc = WMMA_F16(qa1, b1, acc);
      float mk = vmask[lt * 16 + n];      // mask for this lane's column
      float xv = xrow[lt * 16 + n];
#pragma unroll
      for (int v = 0; v < 8; ++v) {
        float s = (mk > 0.5f) ? acc[v] : -1.0e9f;
        float nm = fmaxf(mx[v], s);
        float sc = __expf(mx[v] - nm);    // rescale old state
        float e  = __expf(s - nm);
        se[v] = se[v] * sc + e;
        sx[v] = sx[v] * sc + e * xv;
        mx[v] = nm;
      }
    }
    // cross-lane merge across the 16-lane half (columns) with rescaling
#pragma unroll
    for (int msk = 1; msk <= 8; msk <<= 1) {
#pragma unroll
      for (int v = 0; v < 8; ++v) {
        float mo  = __shfl_xor(mx[v], msk, 32);
        float seo = __shfl_xor(se[v], msk, 32);
        float sxo = __shfl_xor(sx[v], msk, 32);
        float nm  = fmaxf(mx[v], mo);
        float s1  = __expf(mx[v] - nm);
        float s2  = __expf(mo - nm);
        se[v] = se[v] * s1 + seo * s2;
        sx[v] = sx[v] * s1 + sxo * s2;
        mx[v] = nm;
      }
    }
    if (n == 0) {
#pragma unroll
      for (int v = 0; v < 8; ++v) {
        int a = at * 16 + v + 8 * g;
        interp[(h * DM + j) * NALPHA + a] = sx[v] / se[v];
      }
    }
  }
}

// ---------------------------------------------------------------------------
// Kernel 4: e_attn (proj + mean over j), e_imp (conv), gate, output [a][dh]
// ---------------------------------------------------------------------------
__global__ void utde_final(const float* __restrict__ regular,
                           const float* __restrict__ interp,
                           const float* __restrict__ conv_w,
                           const float* __restrict__ conv_b,
                           const float* __restrict__ proj_w,
                           const float* __restrict__ proj_b,
                           const float* __restrict__ g1_w,
                           const float* __restrict__ g1_b,
                           const float* __restrict__ g2_w,
                           const float* __restrict__ g2_b,
                           float* __restrict__ out) {
  __shared__ float regcol[DM];
  __shared__ float S[NH];
  __shared__ float eimp_row[DH], eattn_row[DH], hidden[DH];
  int a = blockIdx.x;
  int t = threadIdx.x;
  if (t < DM) regcol[t] = regular[t * NALPHA + a];
  if (t < NH) {
    float s = 0.0f;
    for (int jj = 0; jj < DM; ++jj) s += interp[(t * DM + jj) * NALPHA + a];
    S[t] = s;
  }
  __syncthreads();
  float ei = conv_b[t];
  for (int m = 0; m < DM; ++m) ei += conv_w[t * DM + m] * regcol[m];
  float ea = 0.0f;
  for (int hh = 0; hh < NH; ++hh) ea += S[hh] * proj_w[t * NH + hh];
  ea = ea * (1.0f / (float)DM) + proj_b[t];
  eimp_row[t] = ei;
  eattn_row[t] = ea;
  __syncthreads();
  float hs = g1_b[t];
  for (int c = 0; c < DH; ++c) hs += eimp_row[c] * g1_w[t * (2 * DH) + c];
  for (int c = 0; c < DH; ++c) hs += eattn_row[c] * g1_w[t * (2 * DH) + DH + c];
  hidden[t] = fmaxf(hs, 0.0f);
  __syncthreads();
  float gs = g2_b[t];
  for (int c = 0; c < DH; ++c) gs += hidden[c] * g2_w[t * DH + c];
  float gg = 1.0f / (1.0f + __expf(-gs));
  out[a * DH + t] = gg * ei + (1.0f - gg) * ea;
}

// ---------------------------------------------------------------------------
extern "C" void kernel_launch(void* const* d_in, const int* in_sizes, int n_in,
                              void* d_out, int out_size, void* d_ws, size_t ws_size,
                              hipStream_t stream) {
  const float* x_ts   = (const float*)d_in[0];
  const float* t_ts   = (const float*)d_in[1];
  const float* gmeans = (const float*)d_in[2];
  const float* conv_w = (const float*)d_in[3];
  const float* conv_b = (const float*)d_in[4];
  const float* t2v_w  = (const float*)d_in[5];
  const float* t2v_phi= (const float*)d_in[6];
  const float* wq_w   = (const float*)d_in[7];
  const float* wq_b   = (const float*)d_in[8];
  const float* wk_w   = (const float*)d_in[9];
  const float* wk_b   = (const float*)d_in[10];
  const float* proj_w = (const float*)d_in[11];
  const float* proj_b = (const float*)d_in[12];
  const float* g1_w   = (const float*)d_in[13];
  const float* g1_b   = (const float*)d_in[14];
  const float* g2_w   = (const float*)d_in[15];
  const float* g2_b   = (const float*)d_in[16];
  float* out = (float*)d_out;

  // Workspace layout: regular (64KB) | q f16 (256KB) | interp (512KB)
  char* ws = (char*)d_ws;
  float*    regular = (float*)ws;                       // DM*NALPHA f32
  _Float16* qf16    = (_Float16*)(ws + 65536);          // NH*NALPHA*DV f16
  float*    interp  = (float*)(ws + 65536 + 262144);    // NH*DM*NALPHA f32

  utde_impute<<<1, 64, 0, stream>>>(x_ts, t_ts, gmeans, regular);
  utde_qprep<<<NH, 256, 0, stream>>>(t2v_w, t2v_phi, wq_w, wq_b, qf16);
  utde_attn<<<dim3(NH, DM), 256, 0, stream>>>(x_ts, t_ts, t2v_w, t2v_phi,
                                              wk_w, wk_b, qf16, interp);
  utde_final<<<NALPHA, 256, 0, stream>>>(regular, interp, conv_w, conv_b,
                                         proj_w, proj_b, g1_w, g1_b,
                                         g2_w, g2_b, out);
}
